// VLSTM_22814866276682
// MI455X (gfx1250) — matrix operand order repaired and checked by reading, not compiled, not run on hardware
//
#include <hip/hip_runtime.h>

// ---------- CDNA5 WMMA types ----------
typedef __attribute__((ext_vector_type(16))) _Float16 v16h;
typedef __attribute__((ext_vector_type(2)))  __fp16   v2fp16;  // return type of cvt_pkrtz
typedef __attribute__((ext_vector_type(8)))  float    v8f;

union ABfrag { v16h v; unsigned u[8]; uint4 q[2]; };

__device__ __forceinline__ unsigned pkf16(float a, float b) {
    union { v2fp16 h; unsigned u; } cv;
    cv.h = __builtin_amdgcn_cvt_pkrtz(a, b);   // v_cvt_pk_rtz_f16_f32
    return cv.u;
}

__device__ __forceinline__ float sigf(float x) { return 1.0f / (1.0f + __expf(-x)); }

// ---------- problem constants ----------
#define TT    20
#define NN    32768
#define DH    128
#define MTILE 32     // nodes per workgroup (2 row-tiles of 16)

// ---------- LDS layout (bytes); total 312736 < 320KB WGP LDS ----------
// Weights stored in WMMA *fragment order*: frag = (kchunk*32 + colTile),
// within a frag lane L owns 8 contiguous dwords (f16 K-pairs) -> ds_load_b128.
#define LDS_WIHF  0                          // 2*32*256 uint = 65536
#define LDS_WHHF  (LDS_WIHF + 2*32*256*4)    // 4*32*256 uint = 131072
#define LDS_BIAS  (LDS_WHHF + 4*32*256*4)    // 512 f32 (b_ih+b_hh)
#define LDS_WE    (LDS_BIAS + 512*4)         // 128 f32
#define LDS_BE    (LDS_WE   + 128*4)         // 64 f32
#define LDS_WOUT  (LDS_BE   + 64*4)          // 5*128 f32 (transposed)
#define LDS_BOUT  (LDS_WOUT + 5*128*4)       // 8 f32
#define LDS_EMB   (LDS_BOUT + 8*4)           // 32*32 uint (f16 pairs)   = 4096
#define LDS_H16   (LDS_EMB  + 32*32*4)       // 32*64 uint (f16 pairs)   = 8192
#define LDS_HF    (LDS_H16  + 32*64*4)       // 32*128 f32               = 16384
#define LDS_C     (LDS_HF   + 32*128*4)      // 32*128 f32               = 16384
#define LDS_G     (LDS_C    + 32*128*4)      // 32*512 f32 (gates; i-slot reused as h_new) = 65536
#define LDS_MASK  (LDS_G    + 32*512*4)      // 32 int
#define LDS_TOTAL (LDS_MASK + 32*4)

#define WMMA_F16(A, B, C) __builtin_amdgcn_wmma_f32_16x16x32_f16( \
        false, (A), false, (B), (short)0, (C), false, false)

__global__ __launch_bounds__(256)
void vlstm_persistent_kernel(const float* __restrict__ nodes,   // (T,N,2)
                             const int*   __restrict__ mask,    // (T,N)
                             const float* __restrict__ h0,      // (N,128)
                             const float* __restrict__ c0,      // (N,128)
                             const float* __restrict__ We,      // (2,64)
                             const float* __restrict__ be,      // (64)
                             const float* __restrict__ Wih,     // (64,512)
                             const float* __restrict__ bih,     // (512)
                             const float* __restrict__ Whh,     // (128,512)
                             const float* __restrict__ bhh,     // (512)
                             const float* __restrict__ Wout,    // (128,5)
                             const float* __restrict__ bout,    // (5)
                             float* __restrict__ out)           // T*N*5 | N*128 | N*128
{
    extern __shared__ char smem[];
    unsigned* sWihF = (unsigned*)(smem + LDS_WIHF);
    unsigned* sWhhF = (unsigned*)(smem + LDS_WHHF);
    float*    sBias = (float*)   (smem + LDS_BIAS);
    float*    sWe   = (float*)   (smem + LDS_WE);
    float*    sbe   = (float*)   (smem + LDS_BE);
    float*    sWout = (float*)   (smem + LDS_WOUT);
    float*    sbout = (float*)   (smem + LDS_BOUT);
    unsigned* sEmb  = (unsigned*)(smem + LDS_EMB);
    unsigned* sH16  = (unsigned*)(smem + LDS_H16);
    float*    sHf   = (float*)   (smem + LDS_HF);
    float*    sC    = (float*)   (smem + LDS_C);
    float*    sG    = (float*)   (smem + LDS_G);
    int*      sMask = (int*)     (smem + LDS_MASK);

    const int tid  = threadIdx.x;
    const int lane = tid & 31;
    const int wave = tid >> 5;
    const int nb   = blockIdx.x * MTILE;   // global node base

    // ---- one-time init: pack weights to f16 K-pair fragments in LDS ----
    // W_ih: K=64 -> 2 K-chunks; frag = kc*32 + colTile; element r: lane=r>>3, j=r&7
    for (int idx = tid; idx < 2 * 32 * 256; idx += 256) {
        int fid = idx >> 8, r = idx & 255;
        int kc = fid >> 5, ct = fid & 31;
        int ln = r >> 3, j = r & 7, hi = ln >> 4;
        int col = ct * 16 + (ln & 15);
        int kp  = kc * 16 + j + hi * 8;               // contiguous K order (same as A): lanes16-31 -> K 16-31
        sWihF[idx] = pkf16(Wih[(2 * kp) * 512 + col], Wih[(2 * kp + 1) * 512 + col]);
    }
    // W_hh: K=128 -> 4 K-chunks
    for (int idx = tid; idx < 4 * 32 * 256; idx += 256) {
        int fid = idx >> 8, r = idx & 255;
        int kc = fid >> 5, ct = fid & 31;
        int ln = r >> 3, j = r & 7, hi = ln >> 4;
        int col = ct * 16 + (ln & 15);
        int kp  = kc * 16 + j + hi * 8;
        sWhhF[idx] = pkf16(Whh[(2 * kp) * 512 + col], Whh[(2 * kp + 1) * 512 + col]);
    }
    for (int c = tid; c < 512; c += 256) sBias[c] = bih[c] + bhh[c];
    if (tid < 128) sWe[tid] = We[tid];
    if (tid < 64)  sbe[tid] = be[tid];
    for (int idx = tid; idx < 5 * 128; idx += 256) {            // W_out transposed [5][128]
        int oc = idx >> 7, k = idx & 127;
        sWout[idx] = Wout[k * 5 + oc];
    }
    if (tid < 5) sbout[tid] = bout[tid];
    for (int idx = tid; idx < MTILE * DH; idx += 256) {         // h, c state
        int n = idx >> 7, j = idx & 127;
        sHf[idx] = h0[(size_t)(nb + n) * DH + j];
        sC[idx]  = c0[(size_t)(nb + n) * DH + j];
    }
    __syncthreads();
    for (int idx = tid; idx < MTILE * 16; idx += 256) {         // pack h -> f16 pairs
        int n = idx >> 4, j0 = (idx & 15) * 8;
        const float* hr = sHf + n * DH + j0;
        unsigned* dst = sH16 + n * (DH / 2) + (j0 >> 1);
        dst[0] = pkf16(hr[0], hr[1]); dst[1] = pkf16(hr[2], hr[3]);
        dst[2] = pkf16(hr[4], hr[5]); dst[3] = pkf16(hr[6], hr[7]);
    }
    __syncthreads();

    // ---- recurrent scan ----
    for (int t = 0; t < TT; ++t) {
        // mask + embedding (K=2 GEMM stays in VALU), packed straight to f16 pairs
        if (tid < MTILE) sMask[tid] = mask[(size_t)t * NN + nb + tid];
        for (int idx = tid; idx < MTILE * 32; idx += 256) {
            int n = idx >> 5, jp = idx & 31, j = jp * 2;
            float x0 = nodes[((size_t)t * NN + nb + n) * 2 + 0];
            float x1 = nodes[((size_t)t * NN + nb + n) * 2 + 1];
            float e0 = fmaxf(0.f, x0 * sWe[j]     + x1 * sWe[64 + j]     + sbe[j]);
            float e1 = fmaxf(0.f, x0 * sWe[j + 1] + x1 * sWe[64 + j + 1] + sbe[j + 1]);
            sEmb[n * 32 + jp] = pkf16(e0, e1);
        }
        __syncthreads();

        // ---- gate GEMM on WMMA ----
        // wave w owns columns [64w,64w+64) as 4 tiles; 2 row-tiles share each B frag.
        {
            const int row16 = lane & 15;
            const int hi    = lane >> 4;          // half-wave (ISA A/B layout)
            ABfrag aE[2][2], aH[2][4];
            #pragma unroll
            for (int rt = 0; rt < 2; ++rt) {
                const int row = rt * 16 + row16;
                #pragma unroll
                for (int kc = 0; kc < 2; ++kc) {  // A = emb (16 x 64 per row-tile)
                    const unsigned* p = sEmb + row * 32 + kc * 16;
                    aE[rt][kc].q[0] = *(const uint4*)(p + hi * 8);
                    aE[rt][kc].q[1] = *(const uint4*)(p + hi * 8 + 4);
                }
                #pragma unroll
                for (int kc = 0; kc < 4; ++kc) {  // A = h (16 x 128 per row-tile)
                    const unsigned* p = sH16 + row * 64 + kc * 16;
                    aH[rt][kc].q[0] = *(const uint4*)(p + hi * 8);
                    aH[rt][kc].q[1] = *(const uint4*)(p + hi * 8 + 4);
                }
            }
            #pragma unroll
            for (int tile = 0; tile < 4; ++tile) {
                const int ct  = wave * 4 + tile;
                const int col = ct * 16 + row16;
                const float bv = sBias[col];
                v8f acc0 = { bv, bv, bv, bv, bv, bv, bv, bv };
                v8f acc1 = { bv, bv, bv, bv, bv, bv, bv, bv };
                #pragma unroll
                for (int kc = 0; kc < 2; ++kc) {  // emb @ W_ih
                    ABfrag b;
                    const unsigned* p = sWihF + (kc * 32 + ct) * 256 + lane * 8;
                    b.q[0] = *(const uint4*)p;
                    b.q[1] = *(const uint4*)(p + 4);
                    acc0 = WMMA_F16(aE[0][kc].v, b.v, acc0);
                    acc1 = WMMA_F16(aE[1][kc].v, b.v, acc1);
                }
                #pragma unroll
                for (int kc = 0; kc < 4; ++kc) {  // h @ W_hh
                    ABfrag b;
                    const unsigned* p = sWhhF + (kc * 32 + ct) * 256 + lane * 8;
                    b.q[0] = *(const uint4*)p;
                    b.q[1] = *(const uint4*)(p + 4);
                    acc0 = WMMA_F16(aH[0][kc].v, b.v, acc0);
                    acc1 = WMMA_F16(aH[1][kc].v, b.v, acc1);
                }
                #pragma unroll
                for (int r = 0; r < 8; ++r) {     // C/D layout: lanes16-31 -> rows +8
                    sG[(r + hi * 8) * 512 + col]      = acc0[r];
                    sG[(16 + r + hi * 8) * 512 + col] = acc1[r];
                }
            }
        }
        __syncthreads();

        // ---- pointwise LSTM cell + masked state update ----
        #pragma unroll
        for (int half = 0; half < 2; ++half) {
            int idx = tid + half * 256;           // 512 units of 8 elements
            int n = idx >> 4, j0 = (idx & 15) * 8;
            int m = sMask[n];
            float hsel[8];
            #pragma unroll
            for (int q = 0; q < 8; ++q) {
                int j = j0 + q;
                float iv = sigf (sG[n * 512 + j]);
                float fv = sigf (sG[n * 512 + 128 + j]);
                float gv = tanhf(sG[n * 512 + 256 + j]);
                float ov = sigf (sG[n * 512 + 384 + j]);
                float cold = sC[n * DH + j];
                float cn = fv * cold + iv * gv;
                float hn = ov * tanhf(cn);
                sG[n * 512 + j] = hn;             // reuse i-gate slot: h_new for W_out
                float cs = m ? cn : cold;
                float hs = m ? hn : sHf[n * DH + j];
                sC[n * DH + j]  = cs;
                sHf[n * DH + j] = hs;
                hsel[q] = hs;
            }
            unsigned* dst = sH16 + n * 64 + (j0 >> 1);   // repack h -> f16 pairs
            dst[0] = pkf16(hsel[0], hsel[1]); dst[1] = pkf16(hsel[2], hsel[3]);
            dst[2] = pkf16(hsel[4], hsel[5]); dst[3] = pkf16(hsel[6], hsel[7]);
        }
        __syncthreads();

        // ---- output projection (N=5: VALU dot-products on h_new in sG) ----
        if (tid < MTILE * 5) {
            int n = tid / 5, oc = tid % 5;
            float s = sbout[oc];
            const float* hr = sG + n * 512;       // h_new (pre-mask)
            const float* w  = sWout + oc * DH;
            #pragma unroll 16
            for (int k = 0; k < DH; ++k) s += hr[k] * w[k];
            out[((size_t)t * NN + nb + n) * 5 + oc] = sMask[n] ? s : 0.0f;
        }
        __syncthreads();
    }

    // ---- final states ----
    {
        const size_t base_h = (size_t)TT * NN * 5;
        const size_t base_c = base_h + (size_t)NN * DH;
        for (int idx = tid; idx < MTILE * DH; idx += 256) {
            int n = idx >> 7, j = idx & 127;
            out[base_h + (size_t)(nb + n) * DH + j] = sHf[idx];
            out[base_c + (size_t)(nb + n) * DH + j] = sC[idx];
        }
    }
}

extern "C" void kernel_launch(void* const* d_in, const int* in_sizes, int n_in,
                              void* d_out, int out_size, void* d_ws, size_t ws_size,
                              hipStream_t stream) {
    (void)in_sizes; (void)n_in; (void)d_ws; (void)ws_size; (void)out_size;
    const float* nodes = (const float*)d_in[0];
    const int*   msk   = (const int*)  d_in[1];
    const float* h0    = (const float*)d_in[2];
    const float* c0    = (const float*)d_in[3];
    const float* We    = (const float*)d_in[4];
    const float* be    = (const float*)d_in[5];
    const float* Wih   = (const float*)d_in[6];
    const float* bih   = (const float*)d_in[7];
    const float* Whh   = (const float*)d_in[8];
    const float* bhh   = (const float*)d_in[9];
    const float* Wout  = (const float*)d_in[10];
    const float* bout  = (const float*)d_in[11];

    // opt in to >64KB dynamic LDS (CDNA5 WGP: 320KB)
    (void)hipFuncSetAttribute((const void*)vlstm_persistent_kernel,
                              hipFuncAttributeMaxDynamicSharedMemorySize,
                              (int)LDS_TOTAL);

    dim3 grid(NN / MTILE), block(256);
    vlstm_persistent_kernel<<<grid, block, LDS_TOTAL, stream>>>(
        nodes, msk, h0, c0, We, be, Wih, bih, Whh, bhh, Wout, bout, (float*)d_out);
}
